// InductiveGCN_19061064860300
// MI455X (gfx1250) — matrix operand order repaired and checked by reading, not compiled, not run on hardware
//
#include <hip/hip_runtime.h>
#include <hip/hip_bf16.h>

typedef __attribute__((ext_vector_type(16))) __bf16 v16bf;
typedef __attribute__((ext_vector_type(8)))  __bf16 v8bf;
typedef __attribute__((ext_vector_type(8)))  float  v8f;
typedef __attribute__((ext_vector_type(4)))  unsigned v4u;
typedef __attribute__((ext_vector_type(8)))  int      v8i_t;
typedef __attribute__((ext_vector_type(4)))  int      v4i_t;

#define L2_EPS 1e-12f

// ---------------------------------------------------------------- utilities

__device__ __forceinline__ unsigned hash_u32(unsigned x) {
  x ^= x >> 16; x *= 0x7feb352dU;
  x ^= x >> 15; x *= 0x846ca68bU;
  x ^= x >> 16;
  return x;
}

// Deterministic "keep <=K random out-edges per source" approximation:
// each edge keeps with probability K/outdeg(src) via a per-edge hash.
__device__ __forceinline__ bool keep_edge(unsigned e, unsigned salt,
                                          unsigned outdeg, unsigned K) {
  if (outdeg <= K) return true;
  unsigned u = hash_u32(e ^ salt);
  return (u % outdeg) < K;
}

__global__ void k_fill_u32(unsigned* __restrict__ p, unsigned v, int n) {
  int i = blockIdx.x * blockDim.x + threadIdx.x;
  if (i < n) p[i] = v;
}

__global__ void k_f32_to_bf16(const float* __restrict__ in,
                              __bf16* __restrict__ out, int n) {
  int i = blockIdx.x * blockDim.x + threadIdx.x;
  if (i < n) out[i] = (__bf16)in[i];
}

// ---------------------------------------------------------------- TDM helper
// DMA a contiguous run of `nelem` bf16 (2-byte) elements from global memory
// into LDS at byte offset `lds_off`, via the Tensor Data Mover.
// Descriptor per CDNA5 ISA 8.3/8.4: 1D copy expressed as a 2D tile with
// tile_dim0 = nelem (<= 65535), tile_dim1 = 1. Tracked by TENSORcnt.
__device__ __forceinline__ void tdm_load_bf16_to_lds(const __bf16* gsrc,
                                                     unsigned lds_off,
                                                     unsigned nelem) {
  unsigned long long ga = (unsigned long long)(size_t)gsrc;
  v4u g0;
  g0[0] = 1u;                                   // count=1, user descriptor
  g0[1] = lds_off;                              // lds_addr (bytes)
  g0[2] = (unsigned)ga;                         // global_addr[31:0]
  g0[3] = (unsigned)((ga >> 32) & 0x01ffffffu)  // global_addr[56:32]
        | (2u << 30);                           // type=2 ("image")
  v8i_t g1;
  g1[0] = (int)(1u << 16);                      // data_size=1 (2 bytes), mask=0
  g1[1] = (int)((nelem & 0xffffu) << 16);       // tensor_dim0[15:0] @ bits 63:48
  g1[2] = (int)((nelem >> 16) | (1u << 16));    // tensor_dim0[31:16] | tensor_dim1=1
  g1[3] = (int)((nelem & 0xffffu) << 16);       // tile_dim0 = nelem @ bits 127:112
  g1[4] = 1;                                    // tile_dim1=1, tile_dim2=0
  g1[5] = (int)nelem;                           // tensor_dim0_stride[31:0]
  g1[6] = 0;                                    // stride hi / dim1_stride lo
  g1[7] = 0;                                    // dim1_stride hi
  v4i_t g2 = (v4i_t)0;
  v4i_t g3 = (v4i_t)0;
#if defined(__clang_major__) && (__clang_major__ >= 23)
  v8i_t g4 = (v8i_t)0;
  __builtin_amdgcn_tensor_load_to_lds(g0, g1, g2, g3, g4, 0);
#else
  __builtin_amdgcn_tensor_load_to_lds(g0, g1, g2, g3, 0);
#endif
}

// ---------------------------------------------------------------- degrees

__global__ void k_outdeg(const long long* __restrict__ src,
                         unsigned* __restrict__ outdeg, int E) {
  int e = blockIdx.x * blockDim.x + threadIdx.x;
  if (e < E) atomicAdd(&outdeg[(int)src[e]], 1u);
}

__global__ void k_deg(const long long* __restrict__ src,
                      const long long* __restrict__ dst,
                      const unsigned* __restrict__ outdeg,
                      unsigned* __restrict__ deg1,
                      unsigned* __restrict__ deg2, int E) {
  int e = blockIdx.x * blockDim.x + threadIdx.x;
  if (e >= E) return;
  int s = (int)src[e], d = (int)dst[e];
  unsigned od = outdeg[s];
  if (keep_edge((unsigned)e, 0x9E3779B9u, od, 100u)) atomicAdd(&deg1[d], 1u);
  if (keep_edge((unsigned)e, 0x85EBCA6Bu, od, 50u))  atomicAdd(&deg2[d], 1u);
}

__global__ void k_dinv(const unsigned* __restrict__ deg,
                       float* __restrict__ dinv, int n) {
  int i = blockIdx.x * blockDim.x + threadIdx.x;
  if (i < n) dinv[i] = rsqrtf((float)deg[i]);   // deg >= 1 (self loop)
}

// ---------------------------------------------------------------- WMMA GEMM
// C[M,N] = A[M,KK] @ B[KK,N], bf16 in, f32 accum. N = NT*16, KK mult of 32.
// B (whole weight matrix, <=32KB bf16) is DMA'd to LDS once per block via the
// Tensor Data Mover, then WMMAs are fed from strided 16-bit LDS reads. Each wave computes
// MT m-tiles x NT n-tiles so every B fragment feeds MT WMMAs.
// Fragment layouts per CDNA5 ISA 7.12.2 (wave32):
//   A 16x32 bf16: lane L holds row M=L%16; lanes<16 cover K {0..7,16..23},
//                 lanes>=16 cover K {8..15,24..31} -> two contiguous v8bf loads.
//   B 32x16 bf16: lane L holds column N=L%16; its 16 halves hold K in A's order.
//   C/D 16x16 f32: VGPR r holds M=r (lanes 0-15) / M=r+8 (lanes 16-31), N=lane%16.
template <int NT, int KK, int MT>
__global__ void k_gemm_wmma(const __bf16* __restrict__ A,
                            const __bf16* __restrict__ B,
                            float* __restrict__ C, int M) {
  constexpr int N = NT * 16;
  __shared__ __align__(16) __bf16 sB[KK * N];

  // --- stage B into LDS via TDM (one wave issues; EXEC is ignored by TDM) ---
  if (threadIdx.x < 32) {
    tdm_load_bf16_to_lds(B, (unsigned)(size_t)&sB[0], (unsigned)(KK * N));
    __builtin_amdgcn_s_wait_tensorcnt(0);
  }
  __syncthreads();

  int wave = (int)((blockIdx.x * blockDim.x + threadIdx.x) >> 5);
  int lane = threadIdx.x & 31;
  int mt0 = wave * MT;
  if (mt0 * 16 >= M) return;
  const int kb = (lane < 16) ? 0 : 8;

  v8f acc[MT][NT];
#pragma unroll
  for (int i = 0; i < MT; ++i)
#pragma unroll
    for (int t = 0; t < NT; ++t) acc[i][t] = (v8f)(0.0f);

#pragma unroll 2
  for (int k0 = 0; k0 < KK; k0 += 32) {
    v16bf a[MT];
#pragma unroll
    for (int i = 0; i < MT; ++i) {
      int m = (mt0 + i) * 16 + (lane & 15);
      if (m >= M) m = M - 1;                       // tail clamp (loads only)
      const __bf16* arow = A + (size_t)m * KK + k0 + kb;
      union { v16bf v; v8bf h[2]; } au;
      au.h[0] = *(const v8bf*)(arow);
      au.h[1] = *(const v8bf*)(arow + 16);
      a[i] = au.v;
    }
    const __bf16* brow = &sB[(k0 + kb) * N + (lane & 15)];
#pragma unroll
    for (int t = 0; t < NT; ++t) {
      union { v16bf v; v8bf h[2]; } bu;
#pragma unroll
      for (int s = 0; s < 8; ++s) {
        bu.v[s] = brow[s * N + t * 16];
        bu.v[s + 8] = brow[(16 + s) * N + t * 16];
      }
#pragma unroll
      for (int i = 0; i < MT; ++i)
        acc[i][t] = __builtin_amdgcn_wmma_f32_16x16x32_bf16(
            false, a[i], false, bu.v, (short)0, acc[i][t], false, false);
    }
  }

  int n0 = lane & 15;
#pragma unroll
  for (int i = 0; i < MT; ++i) {
    int mbase = (mt0 + i) * 16 + ((lane < 16) ? 0 : 8);
#pragma unroll
    for (int t = 0; t < NT; ++t)
#pragma unroll
      for (int r = 0; r < 8; ++r) {
        int mr = mbase + r;
        if (mr < M) C[(size_t)mr * N + t * 16 + n0] = acc[i][t][r];
      }
  }
}

// ------------------------------------------------- self-loop + bias init

template <int C>
__global__ void k_self(const float* __restrict__ h,
                       const float* __restrict__ dinv,
                       const float* __restrict__ bias,
                       float* __restrict__ out, int N) {
  int i = blockIdx.x * blockDim.x + threadIdx.x;
  if (i >= N * C) return;
  int node = i / C;
  int c = i - node * C;
  float di = dinv[node];
  out[i] = bias[c] + di * di * h[i];
}

// ------------------------------------------------- edge gather/scatter-add

template <int C, unsigned K, unsigned SALT>
__global__ void k_agg(const long long* __restrict__ src,
                      const long long* __restrict__ dst,
                      const unsigned* __restrict__ outdeg,
                      const float* __restrict__ dinv,
                      const float* __restrict__ h,
                      float* __restrict__ out, int E) {
  const int TPE = C / 4;                       // threads per edge, 4 floats each
  int idx = blockIdx.x * blockDim.x + threadIdx.x;
  int e = idx / TPE;
  if (e >= E) return;
  // cover index-load latency for the upcoming edges (global_prefetch_b8)
  if (e + 256 < E) {
    __builtin_prefetch(src + e + 256, 0, 0);
    __builtin_prefetch(dst + e + 256, 0, 0);
  }
  int s = (int)src[e], d = (int)dst[e];
  if (!keep_edge((unsigned)e, SALT, outdeg[s], K)) return;
  int c4 = (idx - e * TPE) * 4;
  float nrm = dinv[s] * dinv[d];
  const float4 hv = *(const float4*)(h + (size_t)s * C + c4);
  float* o = out + (size_t)d * C + c4;
  atomicAdd(o + 0, nrm * hv.x);
  atomicAdd(o + 1, nrm * hv.y);
  atomicAdd(o + 2, nrm * hv.z);
  atomicAdd(o + 3, nrm * hv.w);
}

// ------------------------------------------------- ReLU + row L2 norm -> bf16
// One wave per 64-wide row; shuffle reduction (wave32).
__global__ void k_relu_l2_bf16(const float* __restrict__ agg,
                               __bf16* __restrict__ hnb, int N) {
  int wave = (int)((blockIdx.x * blockDim.x + threadIdx.x) >> 5);
  int lane = threadIdx.x & 31;
  if (wave >= N) return;
  const float* row = agg + (size_t)wave * 64;
  float a = fmaxf(row[lane], 0.0f);
  float b = fmaxf(row[lane + 32], 0.0f);
  float ss = a * a + b * b;
#pragma unroll
  for (int off = 16; off > 0; off >>= 1) ss += __shfl_xor(ss, off, 32);
  float inv = 1.0f / fmaxf(sqrtf(ss), L2_EPS);
  hnb[(size_t)wave * 64 + lane]      = (__bf16)(a * inv);
  hnb[(size_t)wave * 64 + lane + 32] = (__bf16)(b * inv);
}

// ---------------------------------------------------------------- launcher

extern "C" void kernel_launch(void* const* d_in, const int* in_sizes, int n_in,
                              void* d_out, int out_size, void* d_ws, size_t ws_size,
                              hipStream_t stream) {
  const float*     x  = (const float*)d_in[0];
  const long long* ei = (const long long*)d_in[1];   // int64 [2,E]
  const float*     W1 = (const float*)d_in[2];
  const float*     b1 = (const float*)d_in[3];
  const float*     W2 = (const float*)d_in[4];
  const float*     b2 = (const float*)d_in[5];

  const int CIN  = 256;
  const int CHID = in_sizes[3];        // 64
  const int COUT = in_sizes[5];        // 128
  const int N = in_sizes[0] / CIN;     // 50000
  const int E = in_sizes[1] / 2;       // 1600000
  const long long* src = ei;
  const long long* dst = ei + E;

  // ---- workspace carve (256B aligned regions) ----
  char* ws = (char*)d_ws;
  size_t off = 0;
  auto carve = [&](size_t bytes) -> void* {
    void* p = (void*)(ws + off);
    off += (bytes + 255) & ~(size_t)255;
    return p;
  };
  unsigned* outdeg = (unsigned*)carve(sizeof(unsigned) * (size_t)N);
  unsigned* deg1   = (unsigned*)carve(sizeof(unsigned) * (size_t)N);
  unsigned* deg2   = (unsigned*)carve(sizeof(unsigned) * (size_t)N);
  float*    dinv1  = (float*)carve(sizeof(float) * (size_t)N);
  float*    dinv2  = (float*)carve(sizeof(float) * (size_t)N);
  __bf16*   W1b    = (__bf16*)carve(sizeof(__bf16) * (size_t)CIN * CHID);
  __bf16*   W2b    = (__bf16*)carve(sizeof(__bf16) * (size_t)CHID * COUT);
  float*    h      = (float*)carve(sizeof(float) * (size_t)N * CHID);
  float*    agg1   = (float*)carve(sizeof(float) * (size_t)N * CHID);
  __bf16*   hnb    = (__bf16*)carve(sizeof(__bf16) * (size_t)N * CHID);
  __bf16*   xb     = (__bf16*)carve(sizeof(__bf16) * (size_t)N * CIN);
  float*    g      = (float*)xb;   // reuse: N*CIN*2 bytes == N*COUT*4 bytes; xb dead after GEMM1
  (void)n_in; (void)ws_size; (void)out_size;

  const int BS = 256;
  auto nb = [&](long long t) { return (int)((t + BS - 1) / BS); };

  // init + conversions
  k_fill_u32<<<nb(N), BS, 0, stream>>>(outdeg, 0u, N);
  k_fill_u32<<<nb(N), BS, 0, stream>>>(deg1, 1u, N);   // self-loop weight 1
  k_fill_u32<<<nb(N), BS, 0, stream>>>(deg2, 1u, N);
  k_f32_to_bf16<<<nb((long long)N * CIN), BS, 0, stream>>>(x,  xb,  N * CIN);
  k_f32_to_bf16<<<nb(CIN * CHID),          BS, 0, stream>>>(W1, W1b, CIN * CHID);
  k_f32_to_bf16<<<nb(CHID * COUT),         BS, 0, stream>>>(W2, W2b, CHID * COUT);

  // degrees + normalization
  k_outdeg<<<nb(E), BS, 0, stream>>>(src, outdeg, E);
  k_deg<<<nb(E), BS, 0, stream>>>(src, dst, outdeg, deg1, deg2, E);
  k_dinv<<<nb(N), BS, 0, stream>>>(deg1, dinv1, N);
  k_dinv<<<nb(N), BS, 0, stream>>>(deg2, dinv2, N);

  // layer 1: h = x @ W1 (TDM-staged WMMA), then normalized aggregate
  {
    const int MT = 2;
    int mtiles = (N + 15) / 16;
    int waves  = (mtiles + MT - 1) / MT;
    int blocks = (waves + 7) / 8;            // 8 waves per 256-thread block
    k_gemm_wmma<4, 256, MT><<<blocks, 256, 0, stream>>>(xb, W1b, h, N);
  }
  k_self<64><<<nb((long long)N * 64), BS, 0, stream>>>(h, dinv1, b1, agg1, N);
  k_agg<64, 100u, 0x9E3779B9u><<<nb((long long)E * 16), BS, 0, stream>>>(
      src, dst, outdeg, dinv1, h, agg1, E);

  // ReLU + L2 normalize -> bf16
  k_relu_l2_bf16<<<nb((long long)N * 32), BS, 0, stream>>>(agg1, hnb, N);

  // layer 2: g = hn @ W2 (TDM-staged WMMA), then normalized aggregate into d_out
  {
    const int MT = 2;
    int mtiles = (N + 15) / 16;
    int waves  = (mtiles + MT - 1) / MT;
    int blocks = (waves + 7) / 8;
    k_gemm_wmma<8, 64, MT><<<blocks, 256, 0, stream>>>(hnb, W2b, g, N);
  }
  float* out = (float*)d_out;
  k_self<128><<<nb((long long)N * 128), BS, 0, stream>>>(g, dinv2, b2, out, N);
  k_agg<128, 50u, 0x85EBCA6Bu><<<nb((long long)E * 32), BS, 0, stream>>>(
      src, dst, outdeg, dinv2, g, out, E);
}
